// BahdanauAttention_69501160784192
// MI455X (gfx1250) — compile-verified, matrix operand order checked by reading, not run on hardware
//
#include <hip/hip_runtime.h>
#include <hip/hip_bf16.h>
#include <math.h>

typedef __attribute__((ext_vector_type(16))) _Float16 v16h;
typedef __attribute__((ext_vector_type(8)))  float    v8f;

// Problem constants (from reference setup_inputs)
constexpr int BB = 64;    // batch
constexpr int SS = 1024;  // sequence
constexpr int EE = 1024;  // encoder dim (K of big GEMM)
constexpr int DD = 512;   // attn dim   (N of big GEMM)

// Workspace layout (bytes)
constexpr size_t BPANEL_OFF = 0;                          // f16, EE*DD*2 = 1 MiB
constexpr size_t HB_OFF     = (size_t)EE * DD * 2;        // f32, BB*DD*4 = 128 KiB
constexpr size_t SC_OFF     = HB_OFF + (size_t)BB*DD*4;   // f32, BB*SS*4 = 256 KiB

// ---------------------------------------------------------------------------
// Kernel 1: hb[b,d] = sum_k hidden[b,k] * attn_W[k,d] + attn_b[d]   (W_h rows)
// ---------------------------------------------------------------------------
__global__ void hproj_kernel(const float* __restrict__ hidden,
                             const float* __restrict__ attn_W,
                             const float* __restrict__ attn_b,
                             float* __restrict__ hb) {
    int b = blockIdx.x, t = threadIdx.x;
    __shared__ float h[DD];
    h[t]       = hidden[b * DD + t];
    h[t + 256] = hidden[b * DD + t + 256];
    __syncthreads();
    for (int d = t; d < DD; d += 256) {
        float s = attn_b[d];
        #pragma unroll 4
        for (int k = 0; k < DD; ++k)
            s += h[k] * attn_W[(size_t)k * DD + d];   // W_h = attn_W[:D]
        hb[(size_t)b * DD + d] = s;
    }
}

// ---------------------------------------------------------------------------
// Kernel 2: swizzle W_e (= attn_W[D:], shape EE x DD, row-major f32) into f16
// B-fragments. Fragment (kc, nt) covers K=kc*32..+31, N=nt*16..+15.
// Lane = n + 16*khalf ; element e -> K = (2*(e/8)+khalf)*8 + (e%8)
// (mirrors ISA 16-bit A 16x32 lane packing, applied to B^T)
// ---------------------------------------------------------------------------
__global__ void bpanel_kernel(const float* __restrict__ attn_W,
                              _Float16* __restrict__ bpanel) {
    int i = blockIdx.x * 256 + threadIdx.x;      // i = k*DD + n, i < EE*DD
    int k = i >> 9;          // / DD
    int n = i & (DD - 1);
    float v = attn_W[(size_t)(DD + k) * DD + n]; // W_e[k][n]
    int kc = k >> 5, kk = k & 31;
    int nt = n >> 4, nn = n & 15;
    int q = kk >> 3, khalf = q & 1, g = q >> 1;
    int lane = nn + (khalf << 4);
    int e    = g * 8 + (kk & 7);
    bpanel[((size_t)(kc * 32 + nt) << 9) + lane * 16 + e] = (_Float16)v;
}

// ---------------------------------------------------------------------------
// Kernel 3: fused  scores[b,s] = v_W . tanh(enc[b,s,:] @ W_e + hb[b,:])
// Block = 256 thr (8 waves), tile = 32 s-rows x 512 N, K loop 32x32.
// Each wave owns 4 N-tiles x 2 M-subtiles -> 8 wmma per K-chunk.
// ---------------------------------------------------------------------------
__global__ void attn_gemm_kernel(const float*   __restrict__ enc,
                                 const _Float16* __restrict__ bpanel,
                                 const float*   __restrict__ hb,
                                 const float*   __restrict__ vW,
                                 float*         __restrict__ scores) {
    const int b  = blockIdx.y;
    const int s0 = blockIdx.x * 32;
    const int t = threadIdx.x, lane = t & 31, wave = t >> 5;

    __shared__ _Float16 lds_a[2 * 512];   // two 16x32 f16 A fragments
    __shared__ float    lds_scores[32];

    if (t < 32) lds_scores[t] = 0.0f;

    v8f acc[4][2];
    #pragma unroll
    for (int j = 0; j < 4; ++j) { acc[j][0] = (v8f){}; acc[j][1] = (v8f){}; }

    const float* aBase = enc + ((size_t)b * SS + s0) * EE;

    for (int kc = 0; kc < EE / 32; ++kc) {
        __syncthreads();
        // cooperative A-tile load: 32 rows x 32 k, f32 -> f16 fragment layout
        {
            int idx = t * 4;                    // 1024 f32 total
            int r = idx >> 5;                   // row 0..31
            int k = idx & 31;                   // k multiple of 4
            const float* src = aBase + (size_t)r * EE + kc * 32 + k;
            float4 v = *(const float4*)src;
            if (kc + 1 < EE / 32)
                __builtin_prefetch(src + 32, 0, 0);   // global_prefetch_b8
            int mf = r >> 4, rr = r & 15;
            int q = k >> 3, khalf = q & 1, g = q >> 1;
            int dl = rr + (khalf << 4);
            int e  = g * 8 + (k & 7);           // k%8 in {0,4} -> e..e+3 contiguous
            _Float16* dst = &lds_a[mf * 512 + dl * 16 + e];
            dst[0] = (_Float16)v.x; dst[1] = (_Float16)v.y;
            dst[2] = (_Float16)v.z; dst[3] = (_Float16)v.w;
        }
        __syncthreads();

        v16h a0 = *(const v16h*)&lds_a[lane * 16];
        v16h a1 = *(const v16h*)&lds_a[512 + lane * 16];

        #pragma unroll
        for (int j = 0; j < 4; ++j) {
            int nt = wave * 4 + j;
            v16h bf = *(const v16h*)(bpanel + ((size_t)(kc * 32 + nt) << 9) + lane * 16);
            acc[j][0] = __builtin_amdgcn_wmma_f32_16x16x32_f16(
                false, a0, false, bf, (short)0, acc[j][0], false, false);
            acc[j][1] = __builtin_amdgcn_wmma_f32_16x16x32_f16(
                false, a1, false, bf, (short)0, acc[j][1], false, false);
        }
    }

    // Epilogue: energy = tanh(c + hb), partial dot with v_W, reduce.
    // C layout: VGPR vr, lanes 0-15 -> M=vr, lanes 16-31 -> M=vr+8; N = lane&15.
    const int n_lo = lane & 15, hi = lane >> 4;
    float p0[8], p1[8];
    #pragma unroll
    for (int vr = 0; vr < 8; ++vr) { p0[vr] = 0.0f; p1[vr] = 0.0f; }

    #pragma unroll
    for (int j = 0; j < 4; ++j) {
        int n = (wave * 4 + j) * 16 + n_lo;
        float hbn = hb[(size_t)b * DD + n];
        float vn  = vW[n];
        #pragma unroll
        for (int vr = 0; vr < 8; ++vr) {
            p0[vr] += vn * tanhf(acc[j][0][vr] + hbn);
            p1[vr] += vn * tanhf(acc[j][1][vr] + hbn);
        }
    }
    // reduce across the 16 lanes sharing `hi` (xor masks 1,2,4,8 stay in-half)
    #pragma unroll
    for (int off = 1; off < 16; off <<= 1) {
        #pragma unroll
        for (int vr = 0; vr < 8; ++vr) {
            p0[vr] += __shfl_xor(p0[vr], off, 32);
            p1[vr] += __shfl_xor(p1[vr], off, 32);
        }
    }
    if (n_lo == 0) {
        #pragma unroll
        for (int vr = 0; vr < 8; ++vr) {
            atomicAdd(&lds_scores[hi * 8 + vr],      p0[vr]);   // rows 0..15
            atomicAdd(&lds_scores[16 + hi * 8 + vr], p1[vr]);   // rows 16..31
        }
    }
    __syncthreads();
    if (t < 32) scores[(size_t)b * SS + s0 + t] = lds_scores[t];
}

// ---------------------------------------------------------------------------
// Kernel 4: masked softmax over S per batch -> attn_weights (second output)
// ---------------------------------------------------------------------------
__global__ void softmax_kernel(const float* __restrict__ scores,
                               const int*   __restrict__ mask,
                               float*       __restrict__ weights) {
    int b = blockIdx.x, t = threadIdx.x, lane = t & 31, wave = t >> 5;
    __shared__ float sc[SS];
    __shared__ float red[8];
    const float NEG = -3.402823466e38f;

    float lm = NEG;
    for (int s = t; s < SS; s += 256) {
        float x = scores[(size_t)b * SS + s];
        if (mask[(size_t)b * SS + s] == 0) x = NEG;
        sc[s] = x;
        lm = fmaxf(lm, x);
    }
    for (int off = 16; off; off >>= 1) lm = fmaxf(lm, __shfl_xor(lm, off, 32));
    if (lane == 0) red[wave] = lm;
    __syncthreads();
    if (t == 0) {
        float m = red[0];
        for (int i = 1; i < 8; ++i) m = fmaxf(m, red[i]);
        red[0] = m;
    }
    __syncthreads();
    float M = red[0];
    __syncthreads();

    float ls = 0.0f;
    for (int s = t; s < SS; s += 256) {
        float p = expf(sc[s] - M);
        sc[s] = p;
        ls += p;
    }
    for (int off = 16; off; off >>= 1) ls += __shfl_xor(ls, off, 32);
    if (lane == 0) red[wave] = ls;
    __syncthreads();
    if (t == 0) {
        float s = 0.0f;
        for (int i = 0; i < 8; ++i) s += red[i];
        red[0] = s;
    }
    __syncthreads();
    float inv = 1.0f / red[0];
    for (int s = t; s < SS; s += 256)
        weights[(size_t)b * SS + s] = sc[s] * inv;
}

// ---------------------------------------------------------------------------
// Kernel 5: context[b,e] = sum_s w[b,s] * enc[b,s,e]   (first output)
// grid (EE/256, BB); weights cached in LDS; coalesced 1KB/row reads.
// ---------------------------------------------------------------------------
__global__ void context_kernel(const float* __restrict__ enc,
                               const float* __restrict__ weights,
                               float*       __restrict__ outctx) {
    int b = blockIdx.y;
    int e = blockIdx.x * 256 + threadIdx.x;
    __shared__ float w[SS];
    for (int s = threadIdx.x; s < SS; s += 256)
        w[s] = weights[(size_t)b * SS + s];
    __syncthreads();
    const float* ebase = enc + (size_t)b * SS * EE + e;
    float acc = 0.0f;
    #pragma unroll 4
    for (int s = 0; s < SS; ++s)
        acc += w[s] * ebase[(size_t)s * EE];
    outctx[(size_t)b * EE + e] = acc;
}

// ---------------------------------------------------------------------------
extern "C" void kernel_launch(void* const* d_in, const int* in_sizes, int n_in,
                              void* d_out, int out_size, void* d_ws, size_t ws_size,
                              hipStream_t stream) {
    const float* hidden = (const float*)d_in[0];
    const float* enc    = (const float*)d_in[1];
    const int*   mask   = (const int*)  d_in[2];
    const float* attn_W = (const float*)d_in[3];
    const float* attn_b = (const float*)d_in[4];
    const float* v_W    = (const float*)d_in[5];

    float* out        = (float*)d_out;
    float* out_ctx    = out;                     // (BB, EE)
    float* out_wts    = out + (size_t)BB * EE;   // (BB, SS)

    char* ws = (char*)d_ws;
    _Float16* bpanel = (_Float16*)(ws + BPANEL_OFF);
    float*    hb     = (float*)(ws + HB_OFF);
    float*    scbuf  = (float*)(ws + SC_OFF);

    (void)in_sizes; (void)n_in; (void)out_size; (void)ws_size;

    hproj_kernel<<<BB, 256, 0, stream>>>(hidden, attn_W, attn_b, hb);
    bpanel_kernel<<<(EE * DD) / 256, 256, 0, stream>>>(attn_W, bpanel);
    attn_gemm_kernel<<<dim3(SS / 32, BB), 256, 0, stream>>>(enc, bpanel, hb, v_W, scbuf);
    softmax_kernel<<<BB, 256, 0, stream>>>(scbuf, mask, out_wts);
    context_kernel<<<dim3(EE / 256, BB), 256, 0, stream>>>(enc, out_wts, out_ctx);
}
